// SimpleLSTMModel_84378927497698
// MI455X (gfx1250) — matrix operand order repaired and checked by reading, not compiled, not run on hardware
//
#include <hip/hip_runtime.h>
#include <hip/hip_bf16.h>
#include <math.h>

typedef __attribute__((ext_vector_type(16))) __bf16 v16bf;
typedef __attribute__((ext_vector_type(8)))  __bf16 v8bf;
typedef __attribute__((ext_vector_type(8)))  float  v8f;
typedef __attribute__((ext_vector_type(4)))  float  v4f;

#define B_TOT 64
#define T_LEN 360
#define FIN   4
#define E_DIM 32
#define C_DIM 128
#define HW    1024
#define H_MAP 32
#define D_DIM 128
#define S_DIM 64
#define A_DIM 64
#define HD    128
#define IF_DIM 128
#define O_DIM 5
#define KCAT  416          // 288 (inp) + 128 (h)
#define NGATES 512
#define BB    16           // batch rows per workgroup

static __device__ inline float sigmoidf_(float x) { return 1.0f / (1.0f + __expf(-x)); }

// WMMA A operand (16x32 bf16) from an LDS row-major [16][stride] buffer.
// Contiguous k layout used here (same as load_b_tile): lane m=lane&15, K range [ (lane>>4)*16, +16 ).
static __device__ inline v16bf load_a_tile(const __bf16* base, int stride, int lane, int k0) {
  int m  = lane & 15;
  int kb = k0 + ((lane >> 4) << 4);
  const __bf16* p = base + m * stride + kb;
  v8bf lo = *(const v8bf*)(p);
  v8bf hi = *(const v8bf*)(p + 8);
  return __builtin_shufflevector(lo, hi, 0,1,2,3,4,5,6,7,8,9,10,11,12,13,14,15);
}

// WMMA B operand (32x16 bf16) from a global row-major [N][K] (transposed-weight) buffer.
// Contiguous k layout used here (same as load_a_tile): lane column n=lane&15; lanes 0-15 K=k0..k0+15, lanes 16-31 K=k0+16..k0+31.
static __device__ inline v16bf load_b_tile(const __bf16* __restrict__ w, int ldk, int lane,
                                           int n0, int k0) {
  int n  = n0 + (lane & 15);
  int kk = k0 + ((lane >> 4) << 4);
  return *(const v16bf*)(w + n * ldk + kk);
}

static __device__ inline v8f wmma_bf16(v16bf a, v16bf b, v8f c) {
  return __builtin_amdgcn_wmma_f32_16x16x32_bf16(false, a, false, b, (short)0, c, false, false);
}

// ---------------- prologue kernels (run once per call) ----------------

__global__ void prep_wcat(const float* __restrict__ W_ih, const float* __restrict__ W_hh,
                          __bf16* __restrict__ wcat_t) {
  int i = blockIdx.x * 256 + threadIdx.x;
  if (i >= NGATES * KCAT) return;
  int n = i / KCAT, k = i % KCAT;
  float v = (k < 288) ? W_ih[k * NGATES + n] : W_hh[(k - 288) * NGATES + n];
  wcat_t[i] = (__bf16)v;
}

__global__ void prep_h2att(const float* __restrict__ W, __bf16* __restrict__ wt) {
  int i = blockIdx.x * 256 + threadIdx.x;
  if (i >= A_DIM * HD) return;
  int a = i / HD, d = i % HD;
  wt[i] = (__bf16)W[d * A_DIM + a];
}

__global__ void prep_aap(const float* __restrict__ W, __bf16* __restrict__ wt) {
  int i = blockIdx.x * 256 + threadIdx.x;
  if (i >= IF_DIM * (HD + D_DIM)) return;
  int o = i / (HD + D_DIM), k = i % (HD + D_DIM);
  wt[i] = (__bf16)W[k * IF_DIM + o];
}

__global__ void prep_penc(const float* __restrict__ fh, const float* __restrict__ Wp,
                          const float* __restrict__ bp, float* __restrict__ p_enc,
                          float* __restrict__ enc_t) {
  int i = blockIdx.x * 256 + threadIdx.x;
  if (i < B_TOT * S_DIM * A_DIM) {
    int b = i >> 12, s = (i >> 6) & 63, a = i & 63;
    float sum = bp[a];
    for (int d = 0; d < D_DIM; ++d) sum += fh[(b * D_DIM + d) * S_DIM + s] * Wp[d * A_DIM + a];
    p_enc[i] = sum;
  }
  if (i < B_TOT * S_DIM * D_DIM) {
    int b = i >> 13, s = (i >> 7) & 63, d = i & 127;
    enc_t[i] = fh[(b * D_DIM + d) * S_DIM + s];
  }
}

// ---------------- persistent scan kernel ----------------

__global__ __launch_bounds__(256, 1) void lstm_att_scan(
    const float* __restrict__ tok, const float* __restrict__ feature,
    const __bf16* __restrict__ wcat_t,
    const float* __restrict__ b_ih, const float* __restrict__ b_hh,
    const float* __restrict__ W_embed, const float* __restrict__ b_embed,
    const __bf16* __restrict__ w_h2att_t, const float* __restrict__ b_h2att,
    const float* __restrict__ W_alpha, const float* __restrict__ b_alpha,
    const float* __restrict__ p_enc, const float* __restrict__ enc_t,
    const __bf16* __restrict__ w_aap_t, const float* __restrict__ b_aap,
    const float* __restrict__ W_out, const float* __restrict__ b_out,
    float* __restrict__ out) {
  __shared__ __align__(16) __bf16 act[BB * KCAT];            // [xemb|in_seq|in_feed|h] bf16
  __shared__ __align__(16) __bf16 gsm[NGATES * BB];          // gates, [n][m]
  __shared__ __align__(16) float  c_s[BB * HD];
  __shared__ __align__(16) float  atth[BB * A_DIM];
  __shared__ __align__(16) float  wgt[BB * S_DIM];
  __shared__ __align__(16) __bf16 aact[BB * (HD + D_DIM)];   // [h|att_res] bf16
  __shared__ __align__(16) float  bias_s[NGATES];            // b_ih + b_hh
  __shared__ __align__(16) float  walpha_s[A_DIM];
  __shared__ __align__(16) float  bh2att_s[A_DIM];
  __shared__ __align__(16) float  baap_s[IF_DIM];
  __shared__ __align__(16) float  wout_s[IF_DIM * O_DIM];
  __shared__ float bout_s[O_DIM + 1];
  __shared__ int idx_s[BB];

  const int tid  = threadIdx.x;
  const int lane = tid & 31;
  const int wave = tid >> 5;
  const int bg   = blockIdx.x * BB;

  // one-time init: zero state, hoist small loop-invariant vectors into LDS
  for (int i = tid; i < BB * KCAT; i += 256) act[i] = (__bf16)0.0f;
  for (int i = tid; i < BB * HD; i += 256) c_s[i] = 0.0f;
  for (int i = tid; i < BB * (HD + D_DIM); i += 256) aact[i] = (__bf16)0.0f;
  for (int i = tid; i < NGATES; i += 256) bias_s[i] = b_ih[i] + b_hh[i];
  if (tid < A_DIM) { walpha_s[tid] = W_alpha[tid]; bh2att_s[tid] = b_h2att[tid]; }
  if (tid < IF_DIM) baap_s[tid] = b_aap[tid];
  for (int i = tid; i < IF_DIM * O_DIM; i += 256) wout_s[i] = W_out[i];
  if (tid < O_DIM) bout_s[tid] = b_out[tid];
  if (tid == O_DIM) bout_s[O_DIM] = b_alpha[0];
  __syncthreads();

#pragma unroll 1
  for (int t = 0; t < T_LEN; ++t) {
    // phase 0: gather index + token embedding
    if (tid < BB) {
      const float* tr = tok + ((size_t)(bg + tid) * T_LEN + t) * FIN;
      int x0 = (int)(tr[0] * 100.0f); if (x0 == -1) x0 = 0;
      int y0 = (int)(tr[1] * 100.0f); if (y0 == -1) y0 = 0;
      idx_s[tid] = y0 * H_MAP + x0;
      if (t + 1 < T_LEN) __builtin_prefetch(tr + FIN, 0, 3);  // warm next step's tokens
    }
#pragma unroll 1
    for (int r = 0; r < 2; ++r) {
      int e = tid + r * 256; int b = e >> 5; int ec = e & 31;
      const float* tr = tok + ((size_t)(bg + b) * T_LEN + t) * FIN;
      float s = b_embed[ec];
#pragma unroll
      for (int f = 0; f < FIN; ++f) s += tr[f] * W_embed[f * E_DIM + ec];
      act[b * KCAT + ec] = (__bf16)s;
    }
    __syncthreads();
    // phase 1: gather encoder feature at token xy
#pragma unroll 1
    for (int r = 0; r < 8; ++r) {
      int e = tid + r * 256; int b = e >> 7; int ch = e & 127;
      float v = feature[((size_t)(bg + b) * C_DIM + ch) * HW + idx_s[b]];
      act[b * KCAT + E_DIM + ch] = (__bf16)v;
    }
    __syncthreads();
    // phase 2: gates GEMM [16,416]x[416,512] — 4 N-tiles/wave, 13 K-steps,
    // 1-deep register double buffer on the B stream (L2-resident weights).
    // unroll 2 -> RA ping-pongs the buffers, eliminating the copy movs.
    {
      v8f acc[4] = {{}, {}, {}, {}};
      v16bf bcur[4], bnxt[4];
#pragma unroll
      for (int n = 0; n < 4; ++n)
        bcur[n] = load_b_tile(wcat_t, KCAT, lane, (wave * 4 + n) * 16, 0);
#pragma unroll 2
      for (int k = 0; k < 13; ++k) {
        int kn = (k + 1 < 13) ? (k + 1) : k;
#pragma unroll
        for (int n = 0; n < 4; ++n)
          bnxt[n] = load_b_tile(wcat_t, KCAT, lane, (wave * 4 + n) * 16, kn * 32);
        v16bf a = load_a_tile(act, KCAT, lane, k * 32);
#pragma unroll
        for (int n = 0; n < 4; ++n) acc[n] = wmma_bf16(a, bcur[n], acc[n]);
#pragma unroll
        for (int n = 0; n < 4; ++n) bcur[n] = bnxt[n];
      }
#pragma unroll
      for (int n = 0; n < 4; ++n) {
        int n_abs = (wave * 4 + n) * 16 + (lane & 15);
        v8bf db;
#pragma unroll
        for (int v = 0; v < 8; ++v) db[v] = (__bf16)acc[n][v];
        *(v8bf*)&gsm[n_abs * BB + ((lane >> 4) << 3)] = db;
      }
    }
    __syncthreads();
    // phase 3: LSTM cell elementwise (torch gate order i,f,g,o)
#pragma unroll 1
    for (int r = 0; r < 8; ++r) {
      int e = tid + r * 256; int b = e >> 7; int j = e & 127;
      float gi = (float)gsm[(j)       * BB + b] + bias_s[j];
      float gf = (float)gsm[(128 + j) * BB + b] + bias_s[128 + j];
      float gg = (float)gsm[(256 + j) * BB + b] + bias_s[256 + j];
      float go = (float)gsm[(384 + j) * BB + b] + bias_s[384 + j];
      float cc = sigmoidf_(gf) * c_s[b * HD + j] + sigmoidf_(gi) * tanhf(gg);
      float hh = sigmoidf_(go) * tanhf(cc);
      c_s[b * HD + j] = cc;
      act[b * KCAT + 288 + j] = (__bf16)hh;
      aact[b * (HD + D_DIM) + j] = (__bf16)hh;
    }
    __syncthreads();
    // phase 4: att_h = h @ W_h2att  (waves 0..3, one 16-col tile each)
    if (wave < 4) {
      v8f acc = {};
      v16bf bcur = load_b_tile(w_h2att_t, HD, lane, wave * 16, 0);
#pragma unroll 2
      for (int k = 0; k < 4; ++k) {
        int kn = (k + 1 < 4) ? (k + 1) : k;
        v16bf bnxt = load_b_tile(w_h2att_t, HD, lane, wave * 16, kn * 32);
        v16bf a = load_a_tile(act, KCAT, lane, 288 + k * 32);
        acc = wmma_bf16(a, bcur, acc);
        bcur = bnxt;
      }
      int n_abs = wave * 16 + (lane & 15);
      float bb = bh2att_s[n_abs];
      int m0 = (lane >> 4) << 3;
#pragma unroll
      for (int v = 0; v < 8; ++v) atth[(m0 + v) * A_DIM + n_abs] = acc[v] + bb;
    }
    __syncthreads();
    // phase 5: scores[b][s] = sum_a tanh(p_enc + att_h) * W_alpha
#pragma unroll 1
    for (int r = 0; r < 4; ++r) {
      int e = tid + r * 256; int b = e >> 6; int s = e & 63;
      const v4f* pp = (const v4f*)(p_enc + ((size_t)(bg + b) * S_DIM + s) * A_DIM);
      const float* ah = atth + b * A_DIM;
      float sc = 0.0f;
#pragma unroll 1
      for (int a4 = 0; a4 < 16; ++a4) {
        v4f p4 = pp[a4];
#pragma unroll
        for (int i = 0; i < 4; ++i) sc += tanhf(p4[i] + ah[a4 * 4 + i]) * walpha_s[a4 * 4 + i];
      }
      wgt[b * S_DIM + s] = sc + bout_s[O_DIM];
    }
    __syncthreads();
    // phase 6: softmax over S (one lane per batch row)
    if (tid < BB) {
      float* w = wgt + tid * S_DIM;
      float mx = w[0];
      for (int s = 1; s < S_DIM; ++s) mx = fmaxf(mx, w[s]);
      float sum = 0.0f;
      for (int s = 0; s < S_DIM; ++s) { float ev = __expf(w[s] - mx); w[s] = ev; sum += ev; }
      float inv = 1.0f / sum;
      for (int s = 0; s < S_DIM; ++s) w[s] *= inv;
    }
    __syncthreads();
    // phase 7: att_res[b][d] = sum_s wgt[b][s] * enc_t[b][s][d]  (per-row B -> VALU float4)
#pragma unroll 1
    for (int r = 0; r < 2; ++r) {
      int slot = tid + r * 256; int b = slot >> 5; int d4 = slot & 31;
      v4f acc = {};
      const float* wrow = wgt + b * S_DIM;
      const v4f* ep = (const v4f*)(enc_t + (size_t)(bg + b) * S_DIM * D_DIM) + d4;
#pragma unroll 4
      for (int s = 0; s < S_DIM; ++s) acc += wrow[s] * ep[s * (D_DIM / 4)];
      int base = b * (HD + D_DIM) + HD + d4 * 4;
#pragma unroll
      for (int i = 0; i < 4; ++i) aact[base + i] = (__bf16)acc[i];
    }
    __syncthreads();
    // phase 8: out = [h|att_res] @ W_aap  (8 waves, one 16-col tile each) -> in_feed
    {
      v8f acc = {};
      v16bf bcur = load_b_tile(w_aap_t, HD + D_DIM, lane, wave * 16, 0);
#pragma unroll 2
      for (int k = 0; k < 8; ++k) {
        int kn = (k + 1 < 8) ? (k + 1) : k;
        v16bf bnxt = load_b_tile(w_aap_t, HD + D_DIM, lane, wave * 16, kn * 32);
        v16bf a = load_a_tile(aact, HD + D_DIM, lane, k * 32);
        acc = wmma_bf16(a, bcur, acc);
        bcur = bnxt;
      }
      int n_abs = wave * 16 + (lane & 15);
      float bb = baap_s[n_abs];
      int m0 = (lane >> 4) << 3;
#pragma unroll
      for (int v = 0; v < 8; ++v) act[(m0 + v) * KCAT + 160 + n_abs] = (__bf16)(acc[v] + bb);
    }
    __syncthreads();
    // phase 9: final projection out5 = out @ W_out + b_out
    if (tid < BB * O_DIM) {
      int b = tid / O_DIM, o = tid % O_DIM;
      float s = bout_s[o];
      const __bf16* fr = act + b * KCAT + 160;
#pragma unroll 4
      for (int i = 0; i < IF_DIM; ++i) s += (float)fr[i] * wout_s[i * O_DIM + o];
      out[((size_t)(bg + b) * T_LEN + t) * O_DIM + o] = s;
    }
    __syncthreads();
  }
}

extern "C" void kernel_launch(void* const* d_in, const int* in_sizes, int n_in,
                              void* d_out, int out_size, void* d_ws, size_t ws_size,
                              hipStream_t stream) {
  const float* tok       = (const float*)d_in[0];
  const float* feature   = (const float*)d_in[1];
  const float* fhid      = (const float*)d_in[2];
  const float* W_embed   = (const float*)d_in[3];
  const float* b_embed   = (const float*)d_in[4];
  const float* W_ih      = (const float*)d_in[5];
  const float* b_ih      = (const float*)d_in[6];
  const float* W_hh      = (const float*)d_in[7];
  const float* b_hh      = (const float*)d_in[8];
  const float* W_h2att   = (const float*)d_in[9];
  const float* b_h2att   = (const float*)d_in[10];
  const float* W_alpha   = (const float*)d_in[11];
  const float* b_alpha   = (const float*)d_in[12];
  const float* W_encproj = (const float*)d_in[13];
  const float* b_encproj = (const float*)d_in[14];
  const float* W_aap     = (const float*)d_in[15];
  const float* b_aap     = (const float*)d_in[16];
  const float* W_out     = (const float*)d_in[17];
  const float* b_out     = (const float*)d_in[18];
  float* out = (float*)d_out;

  char* ws = (char*)d_ws;
  __bf16* wcat_t    = (__bf16*)(ws);                // 512*416*2  = 425984
  __bf16* w_h2att_t = (__bf16*)(ws + 425984);       // 64*128*2   = 16384
  __bf16* w_aap_t   = (__bf16*)(ws + 442368);       // 128*256*2  = 65536
  float*  p_enc     = (float*)(ws + 507904);        // 64*64*64*4 = 1048576
  float*  enc_t     = (float*)(ws + 1556480);       // 64*64*128*4= 2097152

  prep_wcat<<<(NGATES * KCAT + 255) / 256, 256, 0, stream>>>(W_ih, W_hh, wcat_t);
  prep_h2att<<<(A_DIM * HD + 255) / 256, 256, 0, stream>>>(W_h2att, w_h2att_t);
  prep_aap<<<(IF_DIM * (HD + D_DIM) + 255) / 256, 256, 0, stream>>>(W_aap, w_aap_t);
  prep_penc<<<(B_TOT * S_DIM * D_DIM + 255) / 256, 256, 0, stream>>>(fhid, W_encproj, b_encproj,
                                                                     p_enc, enc_t);
  lstm_att_scan<<<B_TOT / BB, 256, 0, stream>>>(
      tok, feature, wcat_t, b_ih, b_hh, W_embed, b_embed, w_h2att_t, b_h2att,
      W_alpha, b_alpha, p_enc, enc_t, w_aap_t, b_aap, W_out, b_out, out);
}
